// KVCachedMultiHeadAttention_88587995447836
// MI455X (gfx1250) — matrix operand order repaired and checked by reading, not compiled, not run on hardware
//
#include <hip/hip_runtime.h>

#define D_MODEL   2048
#define NUM_HEADS 16
#define HEAD_DIM  128
#define NB        4
#define TQL       1024
#define CACHE_T   1024
#define MTOT      (NB * TQL)
#define KTILES    (D_MODEL / 32)

typedef __attribute__((ext_vector_type(16))) __bf16 v16bf;
typedef __attribute__((ext_vector_type(8)))  float  v8f;

union ABFrag { uint4 u4[2]; v16bf v; };

__device__ __forceinline__ unsigned short f2bf_raw(float f) {
  unsigned u = __builtin_bit_cast(unsigned, f);
  u += 0x7FFFu + ((u >> 16) & 1u);          // round-to-nearest-even bf16
  return (unsigned short)(u >> 16);
}
__device__ __forceinline__ unsigned pack2(float a, float b) {
  return (unsigned)f2bf_raw(a) | ((unsigned)f2bf_raw(b) << 16);
}

// ---- CDNA5 async global->LDS copy (ASYNCcnt-tracked, VIMAGE-free DMA path) --
__device__ __forceinline__ void async_b128_0(unsigned lds, unsigned off,
                                             unsigned long long base) {
  asm volatile("global_load_async_to_lds_b128 %0, %1, %2 offset:0"
               :: "v"(lds), "v"(off), "s"(base) : "memory");
}
__device__ __forceinline__ void async_b128_16(unsigned lds, unsigned off,
                                              unsigned long long base) {
  asm volatile("global_load_async_to_lds_b128 %0, %1, %2 offset:16"
               :: "v"(lds), "v"(off), "s"(base) : "memory");
}
__device__ __forceinline__ void wait_async0() {
  asm volatile("s_wait_asynccnt 0" ::: "memory");
}

// ---------------------------------------------------------------------------
// f32 -> bf16 bulk convert (8 elements / thread)
// ---------------------------------------------------------------------------
__global__ __launch_bounds__(256)
void cvt_bf16_kernel(const float* __restrict__ x, unsigned short* __restrict__ y,
                     int n8) {
  const int i = blockIdx.x * 256 + threadIdx.x;
  if (i >= n8) return;
  const float4* p = (const float4*)x + (size_t)i * 2;
  float4 a = p[0], b = p[1];
  uint4 o;
  o.x = pack2(a.x, a.y); o.y = pack2(a.z, a.w);
  o.z = pack2(b.x, b.y); o.w = pack2(b.z, b.w);
  ((uint4*)y)[i] = o;
}

// ---------------------------------------------------------------------------
// Y = A[M,2048] * W[2048,2048]^T + bias  (NT, both row-major along K)
// AMODE 0: A is f32 (cvt during staging). AMODE 1: A is bf16 (async staging).
// outMode 0: f32 row-major. 1: f32 KV-permuted scatter. 2: bf16 row-major.
// Double-buffered LDS, async B staging, preloaded fragments.
// ---------------------------------------------------------------------------
#define GLDA 40
#define GLDB 40

template <int AMODE>
__global__ __launch_bounds__(256)
void gemm2048_nt_wmma(const void* __restrict__ Aany,
                      const unsigned short* __restrict__ Wbf,
                      const float* __restrict__ bias,
                      float* __restrict__ Yf,
                      unsigned short* __restrict__ Ybf,
                      const int outMode) {
  __shared__ __align__(16) unsigned short As[2][128 * GLDA];
  __shared__ __align__(16) unsigned short Bs[2][128 * GLDB];

  const int tid  = threadIdx.x;
  const int wave = tid >> 5;
  const int lane = tid & 31;
  const int ml   = lane & 15;
  const int sel  = lane >> 4;
  const int m0   = blockIdx.y * 128;
  const int n0   = blockIdx.x * 128;

  // staging map: one row per thread, 16 contiguous elements (2 x b128)
  const int srow  = tid >> 1;
  const int shalf = tid & 1;

  const unsigned long long wbase = (unsigned long long)(size_t)Wbf;
  const unsigned ldsB[2] = {
    (unsigned)(size_t)&Bs[0][srow * GLDB + shalf * 16],
    (unsigned)(size_t)&Bs[1][srow * GLDB + shalf * 16] };
  const unsigned bOffBase = (unsigned)(((n0 + srow) * D_MODEL + shalf * 16) * 2);

  const float* xrowF = (const float*)Aany + (size_t)(m0 + srow) * D_MODEL + shalf * 16;
  const unsigned long long abase = (unsigned long long)(size_t)Aany;
  const unsigned ldsA[2] = {
    (unsigned)(size_t)&As[0][srow * GLDA + shalf * 16],
    (unsigned)(size_t)&As[1][srow * GLDA + shalf * 16] };
  const unsigned aOffBase = (unsigned)(((m0 + srow) * D_MODEL + shalf * 16) * 2);

  auto stage = [&](int k0, int buf) {
    const unsigned bo = bOffBase + (unsigned)(k0 * 2);
    async_b128_0(ldsB[buf], bo, wbase);
    async_b128_16(ldsB[buf], bo, wbase);
    if constexpr (AMODE == 1) {
      const unsigned ao = aOffBase + (unsigned)(k0 * 2);
      async_b128_0(ldsA[buf], ao, abase);
      async_b128_16(ldsA[buf], ao, abase);
    } else {
      if (k0 + 64 < D_MODEL) __builtin_prefetch(xrowF + k0 + 64, 0, 3);
      float4 x0 = *(const float4*)(xrowF + k0);
      float4 x1 = *(const float4*)(xrowF + k0 + 4);
      float4 x2 = *(const float4*)(xrowF + k0 + 8);
      float4 x3 = *(const float4*)(xrowF + k0 + 12);
      uint4 pa, pb;
      pa.x = pack2(x0.x, x0.y); pa.y = pack2(x0.z, x0.w);
      pa.z = pack2(x1.x, x1.y); pa.w = pack2(x1.z, x1.w);
      pb.x = pack2(x2.x, x2.y); pb.y = pack2(x2.z, x2.w);
      pb.z = pack2(x3.x, x3.y); pb.w = pack2(x3.z, x3.w);
      *(uint4*)&As[buf][srow * GLDA + shalf * 16]     = pa;
      *(uint4*)&As[buf][srow * GLDA + shalf * 16 + 8] = pb;
    }
  };

  v8f acc[8] = {};

  stage(0, 0);
  wait_async0();
  __syncthreads();

  for (int kt = 0; kt < KTILES; ++kt) {
    const int cur = kt & 1;
    if (kt + 1 < KTILES) stage((kt + 1) * 32, cur ^ 1);

    const unsigned short* Ac = As[cur];
    const unsigned short* Bc = Bs[cur];
    // A fragment: lane ml = M row; dwords 0-3: K=sel*8..+7, 4-7: K=16+sel*8..+7
    ABFrag a;
    a.u4[0] = *(const uint4*)&Ac[(wave * 16 + ml) * GLDA + sel * 8];
    a.u4[1] = *(const uint4*)&Ac[(wave * 16 + ml) * GLDA + 16 + sel * 8];
    // preload all 8 B fragments so WMMAs aren't gated one-by-one on dscnt
    ABFrag bfr[8];
    #pragma unroll
    for (int nt = 0; nt < 8; ++nt) {
      bfr[nt].u4[0] = *(const uint4*)&Bc[(nt * 16 + ml) * GLDB + sel * 8];
      bfr[nt].u4[1] = *(const uint4*)&Bc[(nt * 16 + ml) * GLDB + 16 + sel * 8];
    }
    #pragma unroll
    for (int nt = 0; nt < 8; ++nt)
      acc[nt] = __builtin_amdgcn_wmma_f32_16x16x32_bf16(
          false, a.v, false, bfr[nt].v, (short)0, acc[nt], false, false);

    wait_async0();      // next tile landed in LDS
    __syncthreads();    // visible to all waves; prev buffer free to overwrite
  }

  #pragma unroll
  for (int nt = 0; nt < 8; ++nt) {
    const int n = n0 + nt * 16 + ml;
    const float bv = bias[n];
    #pragma unroll
    for (int i = 0; i < 8; ++i) {
      const int m = m0 + wave * 16 + i + 8 * sel;   // C layout: M = i (+8 upper half)
      const float v = acc[nt][i] + bv;
      if (outMode == 0) {
        Yf[(size_t)m * D_MODEL + n] = v;
      } else if (outMode == 1) {
        const int b = m >> 10;
        const int t = TQL + (m & (TQL - 1));
        const int h = n >> 7;
        const int d = n & 127;
        Yf[(size_t)b * (2048u * 2048u)
           + (size_t)(h * 128 + (t >> 4)) * 2048
           + (size_t)((t & 15) * 128 + d)] = v;
      } else {
        Ybf[(size_t)m * D_MODEL + n] = f2bf_raw(v);
      }
    }
  }
}

// ---------------------------------------------------------------------------
// Flash-style causal attention over the cache (mask kills all j >= 1024).
// Q is bf16 (from proj epilogue); context written bf16 for the O-projection.
// grid: (q-block of 64, head, batch), block = 128 (4 waves, 16 q-rows each).
// ---------------------------------------------------------------------------
#define ALDK 136
#define ALDV 40
#define ALDP 40

__global__ __launch_bounds__(128)
void attn_flash_wmma(const unsigned short* __restrict__ Qbf,
                     const float* __restrict__ CK,
                     const float* __restrict__ CV,
                     unsigned short* __restrict__ CTXbf) {
  __shared__ __align__(16) unsigned short Ks[32 * ALDK];   // [j][d] row-major
  __shared__ __align__(16) unsigned short Vs[128 * ALDV];  // [d][j] transposed
  __shared__ __align__(16) unsigned short Ps[4 * 16 * ALDP];

  const int b    = blockIdx.z;
  const int h    = blockIdx.y;
  const int q0   = blockIdx.x * 64;
  const int tid  = threadIdx.x;
  const int wave = tid >> 5;
  const int lane = tid & 31;
  const int ml   = lane & 15;
  const int sel  = lane >> 4;
  const int qw   = q0 + wave * 16;

  // Q fragment (A layout) straight from bf16 global, 4 chunks of K=32
  ABFrag qf[4];
  const unsigned short* qbase =
      Qbf + (size_t)(b * TQL + qw + ml) * D_MODEL + h * HEAD_DIM;
  #pragma unroll
  for (int c = 0; c < 4; ++c) {
    qf[c].u4[0] = *(const uint4*)(qbase + c * 32 + sel * 8);
    qf[c].u4[1] = *(const uint4*)(qbase + c * 32 + 16 + sel * 8);
  }

  v8f   O[8] = {};
  float mstat[8], lstat[8], alpha[8], rtmp[8];
  #pragma unroll
  for (int i = 0; i < 8; ++i) { mstat[i] = -3.0e38f; lstat[i] = 0.0f; }

  const float scale = 0.088388347648318447f;   // 1/sqrt(128)

  const int srow   = tid >> 2;         // staging: j row 0..31
  const int schunk = (tid & 3) << 5;   // d chunk 0,32,64,96
  const int jEnd   = q0 + 64;          // causal: j <= max row in block

  const float* kbase = CK + (size_t)(b * CACHE_T + srow) * D_MODEL + h * HEAD_DIM + schunk;
  const float* vbase = CV + (size_t)(b * CACHE_T + srow) * D_MODEL + h * HEAD_DIM + schunk;

  for (int j0 = 0; j0 < jEnd; j0 += 32) {
    // ---- stage K (row-major bf16) and V (transposed bf16) tiles ----
    const size_t goff = (size_t)j0 * D_MODEL;
    float kv[32], vv[32];
    #pragma unroll
    for (int q2 = 0; q2 < 8; ++q2) {
      float4 tk = *(const float4*)(kbase + goff + q2 * 4);
      kv[q2*4+0] = tk.x; kv[q2*4+1] = tk.y; kv[q2*4+2] = tk.z; kv[q2*4+3] = tk.w;
      float4 tv = *(const float4*)(vbase + goff + q2 * 4);
      vv[q2*4+0] = tv.x; vv[q2*4+1] = tv.y; vv[q2*4+2] = tv.z; vv[q2*4+3] = tv.w;
    }
    __syncthreads();
    #pragma unroll
    for (int c4 = 0; c4 < 4; ++c4) {
      uint4 p;
      p.x = pack2(kv[c4*8+0], kv[c4*8+1]);
      p.y = pack2(kv[c4*8+2], kv[c4*8+3]);
      p.z = pack2(kv[c4*8+4], kv[c4*8+5]);
      p.w = pack2(kv[c4*8+6], kv[c4*8+7]);
      *(uint4*)&Ks[srow * ALDK + schunk + c4 * 8] = p;
    }
    #pragma unroll
    for (int e = 0; e < 32; ++e)
      Vs[(schunk + e) * ALDV + srow] = f2bf_raw(vv[e]);
    __syncthreads();

    if (j0 <= qw + 15) {   // wave has at least one unmasked column
      // ---- S = Q K^T : two 16x16 n-tiles, K-dim = head_dim in 4 steps ----
      v8f S0 = {}, S1 = {};
      #pragma unroll
      for (int c = 0; c < 4; ++c) {
        ABFrag k0f, k1f;
        k0f.u4[0] = *(const uint4*)&Ks[ml * ALDK + c * 32 + sel * 8];
        k0f.u4[1] = *(const uint4*)&Ks[ml * ALDK + c * 32 + 16 + sel * 8];
        k1f.u4[0] = *(const uint4*)&Ks[(ml + 16) * ALDK + c * 32 + sel * 8];
        k1f.u4[1] = *(const uint4*)&Ks[(ml + 16) * ALDK + c * 32 + 16 + sel * 8];
        S0 = __builtin_amdgcn_wmma_f32_16x16x32_bf16(
            false, qf[c].v, false, k0f.v, (short)0, S0, false, false);
        S1 = __builtin_amdgcn_wmma_f32_16x16x32_bf16(
            false, qf[c].v, false, k1f.v, (short)0, S1, false, false);
      }
      // ---- scale + causal mask + online softmax (f32) ----
      const int rowb = qw + 8 * sel;
      const int col0 = j0 + ml;
      const int col1 = col0 + 16;
      #pragma unroll
      for (int i = 0; i < 8; ++i) {
        float s0 = S0[i] * scale;
        float s1 = S1[i] * scale;
        if (col0 > rowb + i) s0 = -3.0e38f;
        if (col1 > rowb + i) s1 = -3.0e38f;
        S0[i] = s0; S1[i] = s1;
        rtmp[i] = fmaxf(s0, s1);
      }
      #pragma unroll
      for (int off = 8; off >= 1; off >>= 1)
        #pragma unroll
        for (int i = 0; i < 8; ++i)
          rtmp[i] = fmaxf(rtmp[i], __shfl_xor(rtmp[i], off, 32));
      #pragma unroll
      for (int i = 0; i < 8; ++i) {
        const float nm = fmaxf(mstat[i], rtmp[i]);
        alpha[i] = __expf(mstat[i] - nm);
        mstat[i] = nm;
        const float p0 = __expf(S0[i] - nm);
        const float p1 = __expf(S1[i] - nm);
        S0[i] = p0; S1[i] = p1;
        rtmp[i] = p0 + p1;
      }
      #pragma unroll
      for (int off = 8; off >= 1; off >>= 1)
        #pragma unroll
        for (int i = 0; i < 8; ++i)
          rtmp[i] += __shfl_xor(rtmp[i], off, 32);
      #pragma unroll
      for (int i = 0; i < 8; ++i)
        lstat[i] = lstat[i] * alpha[i] + rtmp[i];
      #pragma unroll
      for (int nt = 0; nt < 8; ++nt)
        #pragma unroll
        for (int i = 0; i < 8; ++i)
          O[nt][i] *= alpha[i];

      // ---- P (C layout) -> LDS -> A layout; same-wave LDS is in-order ----
      #pragma unroll
      for (int i = 0; i < 8; ++i) {
        Ps[(wave * 16 + 8 * sel + i) * ALDP + ml]      = f2bf_raw(S0[i]);
        Ps[(wave * 16 + 8 * sel + i) * ALDP + ml + 16] = f2bf_raw(S1[i]);
      }
      ABFrag pf;
      pf.u4[0] = *(const uint4*)&Ps[(wave * 16 + ml) * ALDP + sel * 8];
      pf.u4[1] = *(const uint4*)&Ps[(wave * 16 + ml) * ALDP + 16 + sel * 8];
      // ---- O += P V : 8 d-tiles, K-dim = 32 j's ----
      #pragma unroll
      for (int nt = 0; nt < 8; ++nt) {
        ABFrag vf;
        vf.u4[0] = *(const uint4*)&Vs[(nt * 16 + ml) * ALDV + sel * 8];
        vf.u4[1] = *(const uint4*)&Vs[(nt * 16 + ml) * ALDV + 16 + sel * 8];
        O[nt] = __builtin_amdgcn_wmma_f32_16x16x32_bf16(
            false, pf.v, false, vf.v, (short)0, O[nt], false, false);
      }
    }
  }

  #pragma unroll
  for (int nt = 0; nt < 8; ++nt) {
    #pragma unroll
    for (int i = 0; i < 8; ++i) {
      const int m = qw + i + 8 * sel;
      const float denom = (lstat[i] > 0.0f) ? lstat[i] : 1.0f;
      CTXbf[(size_t)(b * TQL + m) * D_MODEL + h * HEAD_DIM + nt * 16 + ml] =
          f2bf_raw(O[nt][i] / denom);
    }
  }
}

// ---------------------------------------------------------------------------
// cache K/V -> new_k/new_v head-major permuted rows (t < 1024 half)
// ---------------------------------------------------------------------------
__global__ __launch_bounds__(256)
void permute_cache_kernel(const float* __restrict__ CK,
                          const float* __restrict__ CV,
                          float* __restrict__ NK,
                          float* __restrict__ NV) {
  const size_t idx4 = (size_t)blockIdx.x * 256 + threadIdx.x;
  const size_t total4 = (size_t)NB * CACHE_T * D_MODEL / 4;
  if (idx4 >= total4) return;
  const size_t e = idx4 << 2;
  const int b   = (int)(e >> 21);
  const int rem = (int)(e & ((1u << 21) - 1));
  const int t   = rem >> 11;
  const int c   = rem & 2047;
  const int h   = c >> 7;
  const int d   = c & 127;
  const size_t dst = (size_t)b * (2048u * 2048u)
                   + (size_t)(h * 128 + (t >> 4)) * 2048
                   + (size_t)((t & 15) * 128 + d);
  *(float4*)&NK[dst] = *(const float4*)&CK[e];
  *(float4*)&NV[dst] = *(const float4*)&CV[e];
}

extern "C" void kernel_launch(void* const* d_in, const int* in_sizes, int n_in,
                              void* d_out, int out_size, void* d_ws, size_t ws_size,
                              hipStream_t stream) {
  const float* query  = (const float*)d_in[0];
  const float* key    = (const float*)d_in[1];
  const float* value  = (const float*)d_in[2];
  const float* cacheK = (const float*)d_in[3];
  const float* cacheV = (const float*)d_in[4];
  const float* Wq = (const float*)d_in[5];  const float* bq = (const float*)d_in[6];
  const float* Wk = (const float*)d_in[7];  const float* bk = (const float*)d_in[8];
  const float* Wv = (const float*)d_in[9];  const float* bv = (const float*)d_in[10];
  const float* Wo = (const float*)d_in[11]; const float* bo = (const float*)d_in[12];

  float* out   = (float*)d_out;
  float* new_k = out + (size_t)NB * TQL * D_MODEL;        // 8,388,608
  float* new_v = new_k + (size_t)NB * 2048 * D_MODEL;     // +16,777,216

  // ws (bf16 units): [Wq|Wk|Wv|Wo bf16][Q bf16]; ctx aliases dead Wq+Wk region
  unsigned short* wsbf  = (unsigned short*)d_ws;
  unsigned short* wq_bf = wsbf;
  unsigned short* wk_bf = wq_bf + (size_t)D_MODEL * D_MODEL;
  unsigned short* wv_bf = wk_bf + (size_t)D_MODEL * D_MODEL;
  unsigned short* wo_bf = wv_bf + (size_t)D_MODEL * D_MODEL;
  unsigned short* q_bf  = wo_bf + (size_t)D_MODEL * D_MODEL;
  unsigned short* ctx_bf = wsbf;   // 8,388,608 elems over Wq+Wk (dead by then)

  const dim3 gemmGrid(D_MODEL / 128, MTOT / 128);
  const int wN8 = D_MODEL * D_MODEL / 8;

  permute_cache_kernel<<<(NB * CACHE_T * D_MODEL / 4 + 255) / 256, 256, 0, stream>>>(
      cacheK, cacheV, new_k, new_v);

  cvt_bf16_kernel<<<(wN8 + 255) / 256, 256, 0, stream>>>(Wq, wq_bf, wN8);
  cvt_bf16_kernel<<<(wN8 + 255) / 256, 256, 0, stream>>>(Wk, wk_bf, wN8);
  cvt_bf16_kernel<<<(wN8 + 255) / 256, 256, 0, stream>>>(Wv, wv_bf, wN8);
  cvt_bf16_kernel<<<(wN8 + 255) / 256, 256, 0, stream>>>(Wo, wo_bf, wN8);

  gemm2048_nt_wmma<0><<<gemmGrid, 256, 0, stream>>>(query, wq_bf, bq,
                                                    nullptr, q_bf, 2);
  gemm2048_nt_wmma<0><<<gemmGrid, 256, 0, stream>>>(key,   wk_bf, bk,
                                                    new_k, nullptr, 1);
  gemm2048_nt_wmma<0><<<gemmGrid, 256, 0, stream>>>(value, wv_bf, bv,
                                                    new_v, nullptr, 1);
  attn_flash_wmma<<<dim3(TQL / 64, NUM_HEADS, NB), 128, 0, stream>>>(
      q_bf, cacheK, cacheV, ctx_bf);
  gemm2048_nt_wmma<1><<<gemmGrid, 256, 0, stream>>>(ctx_bf, wo_bf, bo,
                                                    out, nullptr, 0);

  (void)in_sizes; (void)n_in; (void)out_size; (void)ws_size;
}
